// cnn_43662637531913
// MI455X (gfx1250) — matrix operand order repaired and checked by reading, not compiled, not run on hardware
//
#include <hip/hip_runtime.h>
#include <hip/hip_bf16.h>

typedef _Float16 half_t;
typedef __attribute__((ext_vector_type(16))) _Float16 v16h;
typedef __attribute__((ext_vector_type(8)))  float    v8f;

// ---------------------------------------------------------------------------
// WMMA helpers (CDNA5 gfx1250, wave32)
// ---------------------------------------------------------------------------
static __device__ __forceinline__ v8f wmma_f16(v16h a, v16h b, v8f c) {
  // D = A(16x32 f16) x B(32x16 f16) + C(16x16 f32)
  return __builtin_amdgcn_wmma_f32_16x16x32_f16(
      /*neg_a=*/false, a, /*neg_b=*/false, b,
      /*c_mod=*/(short)0, c, /*reuse_a=*/false, /*reuse_b=*/false);
}

// A fragment (16x32, f16) from row-major source with row stride `ld` (halfs).
// Layout: lanes 0-15 -> M=lane, K 0..7 (+16..23); lanes 16-31 -> K 8..15 (+24..31)
static __device__ __forceinline__ v16h load_a_frag(const half_t* src, int ld) {
  const int lane = threadIdx.x & 31;
  const half_t* p = src + (lane & 15) * ld + ((lane >> 4) << 3);
  v16h a;
#pragma unroll
  for (int i = 0; i < 8; ++i) a[i] = p[i];
#pragma unroll
  for (int i = 0; i < 8; ++i) a[8 + i] = p[16 + i];
  return a;
}

// B fragment (32x16, f16) built from a row-major fp32 weight matrix W[G][ldk]:
// tile covers gate/output columns g0..g0+15 and k rows k0..k0+31.
// B[k][n] = W[g0+n][k]  -> each lane reads two 8-f32 runs (K 8h..8h+7, 16+8h..), cvt to f16.
static __device__ __forceinline__ v16h load_b_frag_f32(const float* W, int ldk,
                                                       int g0, int k0) {
  const int lane = threadIdx.x & 31;
  const float* p = W + (long)(g0 + (lane & 15)) * ldk + k0 + ((lane >> 4) << 3);
  v16h b;
#pragma unroll
  for (int i = 0; i < 8; ++i) b[i] = (half_t)p[i];
#pragma unroll
  for (int i = 0; i < 8; ++i) b[8 + i] = (half_t)p[16 + i];
  return b;
}

// Branchless gate math on the serial critical path: single v_exp_f32 +
// v_rcp_f32, no OCML exec-mask branches.
static __device__ __forceinline__ float sigm(float x) {
  return __builtin_amdgcn_rcpf(1.0f + __expf(-x));
}
static __device__ __forceinline__ float tanh_fast(float x) {
  // tanh(x) = 2*sigmoid(2x) - 1, saturates cleanly at +-1
  return 2.0f * __builtin_amdgcn_rcpf(1.0f + __expf(-2.0f * x)) - 1.0f;
}

// ---------------------------------------------------------------------------
// zero fill
// ---------------------------------------------------------------------------
__global__ void k_zero(float* p, long n) {
  long i = (long)blockIdx.x * blockDim.x + threadIdx.x;
  long stride = (long)gridDim.x * blockDim.x;
  for (; i < n; i += stride) p[i] = 0.0f;
}

// ---------------------------------------------------------------------------
// LSTM1 layer body, specialized on input width (32 for layer 0, 64 after).
// Straight-line WMMA chains, no runtime K branches in the scan loop.
// ---------------------------------------------------------------------------
template <int KIN>
static __device__ __forceinline__ void lstm1_layer(
    const float* __restrict__ Wih, const float* __restrict__ Whh,
    const float* __restrict__ bb, half_t (*buf)[20][64], half_t (*hbuf)[16][64],
    int tid, int w, int nn, int hl, v8f& hsum, v8f& csum) {
  constexpr int NKS = KIN / 32;

  // loop-invariant weight fragments in registers (4 gates x k-steps)
  v16h wihB[4][NKS];
  v16h whhB[4][2];
  v8f  biasF[4];
#pragma unroll
  for (int q = 0; q < 4; ++q) {
    const int g0 = q * 64 + w * 16;
#pragma unroll
    for (int ks = 0; ks < NKS; ++ks)
      wihB[q][ks] = load_b_frag_f32(Wih, KIN, g0, ks * 32);
    whhB[q][0] = load_b_frag_f32(Whh, 64, g0, 0);
    whhB[q][1] = load_b_frag_f32(Whh, 64, g0, 32);
    const float bv = bb[g0 + nn];
#pragma unroll
    for (int r = 0; r < 8; ++r) biasF[q][r] = bv;
  }

  // h0 = c0 = ones (reference init)
  for (int i = tid; i < 16 * 64; i += 128) hbuf[0][i >> 6][i & 63] = (half_t)1.0f;
  v8f c;
#pragma unroll
  for (int r = 0; r < 8; ++r) c[r] = 1.0f;
  __syncthreads();

  int cur = 0;
  v8f hlast = {};
  for (int t = 0; t < 20; ++t) {
    v16h ax[NKS];
#pragma unroll
    for (int ks = 0; ks < NKS; ++ks)
      ax[ks] = load_a_frag(&buf[0][t][ks * 32], 20 * 64);
    v16h ah0 = load_a_frag(&hbuf[cur][0][0], 64);
    v16h ah1 = load_a_frag(&hbuf[cur][0][32], 64);

    v8f acc[4];
#pragma unroll
    for (int q = 0; q < 4; ++q) {
      v8f a = biasF[q];
#pragma unroll
      for (int ks = 0; ks < NKS; ++ks) a = wmma_f16(ax[ks], wihB[q][ks], a);
      a = wmma_f16(ah0, whhB[q][0], a);
      a = wmma_f16(ah1, whhB[q][1], a);
      acc[q] = a;
    }
    // gates (torch order i,f,g,o), branchless
#pragma unroll
    for (int r = 0; r < 8; ++r) {
      float ig = sigm(acc[0][r]);
      float fg = sigm(acc[1][r]);
      float gg = tanh_fast(acc[2][r]);
      float og = sigm(acc[3][r]);
      float cn = fg * c[r] + ig * gg;
      c[r] = cn;
      hlast[r] = og * tanh_fast(cn);
    }
    __syncthreads();  // all waves done reading buf[.][t] (rewritten in place)
    const int nxt = cur ^ 1;
#pragma unroll
    for (int r = 0; r < 8; ++r) {
      const int M = r + hl * 8;
      const half_t hv = (half_t)hlast[r];
      hbuf[nxt][M][w * 16 + nn] = hv;
      buf[M][t][w * 16 + nn] = hv;  // next layer's input, in place
    }
    __syncthreads();
    cur = nxt;
  }
#pragma unroll
  for (int r = 0; r < 8; ++r) {
    hsum[r] += hlast[r];
    csum[r] += c[r];
  }
}

// ---------------------------------------------------------------------------
// LSTM1: fused embedding + 4-layer LSTM(h=64) over 16 sequences per block.
// Block = 128 threads (4 waves). Wave w owns hidden units [16w,16w+16).
// Activations (16 x 20 x 64 f16) live in LDS and are rewritten in place
// layer-by-layer. Final (sum_h + sum_c)/4 scattered to padded[B][1024][64].
// ---------------------------------------------------------------------------
__global__ void __launch_bounds__(128) k_lstm1(
    const int* __restrict__ atom_ids, const float* __restrict__ embed_w,
    const float* __restrict__ Wih0, const float* __restrict__ Whh0,
    const float* __restrict__ b0,
    const float* __restrict__ WihS, const float* __restrict__ WhhS,
    const float* __restrict__ bS,
    const int* __restrict__ batch_idx, const int* __restrict__ pos_idx,
    float* __restrict__ padded) {
  __shared__ __align__(16) half_t buf[16][20][64];   // activations (in-place)
  __shared__ __align__(16) half_t hbuf[2][16][64];   // h ping-pong

  const int tid = threadIdx.x;
  const int w = tid >> 5;
  const int lane = tid & 31;
  const int nn = lane & 15;
  const int hl = lane >> 4;
  const long n0 = (long)blockIdx.x * 16;

  // --- embedding: x = embed_w[id] * id, f16 into buf cols 0..31 ---
  for (int i = tid; i < 16 * 20 * 32; i += 128) {
    int s = i / (20 * 32);
    int r = i % (20 * 32);
    int t = r / 32;
    int e = r % 32;
    int id = atom_ids[(n0 + s) * 20 + t];
    buf[s][t][e] = (half_t)(embed_w[id * 32 + e] * (float)id);
  }
  __syncthreads();

  v8f hsum = {};
  v8f csum = {};

  lstm1_layer<32>(Wih0, Whh0, b0, buf, hbuf, tid, w, nn, hl, hsum, csum);
#pragma unroll 1
  for (int l = 1; l < 4; ++l) {
    lstm1_layer<64>(WihS + (long)(l - 1) * 256 * 64,
                    WhhS + (long)(l - 1) * 256 * 64,
                    bS + (long)(l - 1) * 256, buf, hbuf, tid, w, nn, hl, hsum,
                    csum);
  }

  // scatter d = (mean_h + mean_c) into padded[b][p][:]
#pragma unroll
  for (int r = 0; r < 8; ++r) {
    const int M = r + hl * 8;
    const long sg = n0 + M;
    const int bi = batch_idx[sg];
    const int pi = pos_idx[sg];
    padded[((long)bi * 1024 + pi) * 64 + w * 16 + nn] = (hsum[r] + csum[r]) * 0.25f;
  }
}

// ---------------------------------------------------------------------------
// BN over (1024*64) per structure, training-mode biased var; emit f16 x2.
// ---------------------------------------------------------------------------
__global__ void k_bn2(const float* __restrict__ padded, const float* __restrict__ g,
                      const float* __restrict__ b, half_t* __restrict__ x2) {
  const int bi = blockIdx.x;
  const int tid = threadIdx.x;  // 256
  const float* src = padded + (long)bi * 65536;
  float s = 0.f, s2 = 0.f;
  for (int i = tid; i < 65536; i += 256) {
    float v = src[i];
    s += v;
    s2 += v * v;
  }
  __shared__ float rs[256], rq[256];
  rs[tid] = s;
  rq[tid] = s2;
  __syncthreads();
  for (int o = 128; o > 0; o >>= 1) {
    if (tid < o) { rs[tid] += rs[tid + o]; rq[tid] += rq[tid + o]; }
    __syncthreads();
  }
  const float m = rs[0] * (1.0f / 65536.0f);
  const float var = rq[0] * (1.0f / 65536.0f) - m * m;
  const float sc = rsqrtf(var + 1e-5f) * g[bi];
  const float sh = b[bi];
  half_t* dst = x2 + (long)bi * 65536;
  for (int i = tid; i < 65536; i += 256) dst[i] = (half_t)((src[i] - m) * sc + sh);
}

// ---------------------------------------------------------------------------
// LSTM2 input GEMM: xgT[t][col][b] = (inp @ Wih^T + bias)  (transposed store:
// per-lane contiguous 8-float runs, matching the WMMA C/D register layout).
// Row index = t*32 + b; addr(inp) = b*bstr + t*tstr (halfs). 2048 blocks x 128.
// ---------------------------------------------------------------------------
template <int KIN>
__global__ void __launch_bounds__(128) k_gemm_xg(
    const half_t* __restrict__ A, long tstr, long bstr,
    const float* __restrict__ Wih, const float* __restrict__ bias,
    float* __restrict__ xgT) {
  constexpr int NKS = KIN / 32;
  const int tid = threadIdx.x;
  const int wv = tid >> 5;
  const int lane = tid & 31;
  const int nn = lane & 15;
  const int hl = lane >> 4;
  const long row0 = (long)blockIdx.x * 16;
  const long t0 = row0 >> 5;        // 16-row tile lies within one t (32 rows/t)
  const int bbase = (int)(row0 & 31) + hl * 8;

  // A fragments for this 16-row tile
  v16h a[NKS];
  {
    const long row = row0 + nn;
    const long tt = row >> 5;
    const long bb = row & 31;
    const half_t* base = A + bb * bstr + tt * tstr + (hl << 3);
#pragma unroll
    for (int ks = 0; ks < NKS; ++ks) {
      const half_t* p = base + ks * 32;
#pragma unroll
      for (int i = 0; i < 8; ++i) a[ks][i] = p[i];
#pragma unroll
      for (int i = 0; i < 8; ++i) a[ks][8 + i] = p[16 + i];
    }
  }
#pragma unroll
  for (int j = 0; j < 8; ++j) {
    const int g0 = (wv * 8 + j) * 16;
    v8f acc;
    const float bv = bias[g0 + nn];
#pragma unroll
    for (int r = 0; r < 8; ++r) acc[r] = bv;
#pragma unroll
    for (int ks = 0; ks < NKS; ++ks)
      acc = wmma_f16(a[ks], load_b_frag_f32(Wih, KIN, g0, ks * 32), acc);
    // transposed store: xgT[(t*512 + col)*32 + b], 8 consecutive floats/lane
    float* dst = xgT + (t0 * 512 + g0 + nn) * 32 + bbase;
#pragma unroll
    for (int r = 0; r < 8; ++r) dst[r] = acc[r];
  }
}

// ---------------------------------------------------------------------------
// LSTM2 recurrence: 1 block x 512 threads (16 waves), 1024 serial steps.
// Wave (m = wv>>3, s = wv&7): rows [16m,16m+16), units [16s,16s+16).
// Whh fragments (4 gates x 4 k-steps) resident in registers; xgT reads are
// per-lane contiguous (C-fragment layout). ONE barrier per step: h ping-pong
// makes the pre-write barrier unnecessary (post-write barrier of step t
// orders all step-t reads before any step-t+1 write of the same buffer).
// ---------------------------------------------------------------------------
__global__ void __launch_bounds__(512) k_lstm2_rec(
    const float* __restrict__ xgT,  // [1024][512][32] with bias folded in
    const float* __restrict__ Whh,  // [512][128]
    half_t* __restrict__ hout,      // [1024][32][128]
    float* __restrict__ csum,       // [32][128] accumulated across layers
    int write_h) {                  // 0 on the last layer (h never consumed)
  __shared__ __align__(16) half_t hl_[2][32][128];
  const int tid = threadIdx.x;
  const int wv = tid >> 5;
  const int lane = tid & 31;
  const int nn = lane & 15;
  const int hl = lane >> 4;
  const int m = wv >> 3;
  const int s = wv & 7;
  const int brow = m * 16 + hl * 8;

  v16h whhB[4][4];
#pragma unroll
  for (int q = 0; q < 4; ++q)
#pragma unroll
    for (int ks = 0; ks < 4; ++ks)
      whhB[q][ks] = load_b_frag_f32(Whh, 128, q * 128 + s * 16, ks * 32);

  v8f c;
#pragma unroll
  for (int r = 0; r < 8; ++r) c[r] = 1.0f;
  for (int i = tid; i < 32 * 128; i += 512) hl_[0][i >> 7][i & 127] = (half_t)1.0f;
  __syncthreads();

  int cur = 0;
  for (int t = 0; t < 1024; ++t) {
    v16h ah[4];
#pragma unroll
    for (int ks = 0; ks < 4; ++ks)
      ah[ks] = load_a_frag(&hl_[cur][m * 16][ks * 32], 128);

    const float* xr = xgT + (long)t * 512 * 32;
    if (t + 1 < 1024)
      __builtin_prefetch(xr + 512 * 32 + (s * 16 + nn) * 32 + brow, 0, 0);

    v8f acc[4];
#pragma unroll
    for (int q = 0; q < 4; ++q) {
      v8f a;
      const float* xp = xr + (q * 128 + s * 16 + nn) * 32 + brow;
#pragma unroll
      for (int r = 0; r < 8; ++r) a[r] = xp[r];
#pragma unroll
      for (int ks = 0; ks < 4; ++ks) a = wmma_f16(ah[ks], whhB[q][ks], a);
      acc[q] = a;
    }
    v8f hv;
#pragma unroll
    for (int r = 0; r < 8; ++r) {
      float cn = sigm(acc[1][r]) * c[r] + sigm(acc[0][r]) * tanh_fast(acc[2][r]);
      c[r] = cn;
      hv[r] = sigm(acc[3][r]) * tanh_fast(cn);
    }
    const int nxt = cur ^ 1;
#pragma unroll
    for (int r = 0; r < 8; ++r) {
      const int row = m * 16 + r + hl * 8;
      hl_[nxt][row][s * 16 + nn] = (half_t)hv[r];
    }
    if (write_h) {  // uniform scalar branch
#pragma unroll
      for (int r = 0; r < 8; ++r) {
        const int row = m * 16 + r + hl * 8;
        hout[((long)t * 32 + row) * 128 + s * 16 + nn] = (half_t)hv[r];
      }
    }
    __syncthreads();
    cur = nxt;
  }
#pragma unroll
  for (int r = 0; r < 8; ++r) {
    const int row = m * 16 + r + hl * 8;
    csum[row * 128 + s * 16 + nn] += c[r];
  }
}

// ---------------------------------------------------------------------------
// Head: mean cT2, row-BN, MLP chain, relu(sum). 1 block x 128 threads.
// ---------------------------------------------------------------------------
__device__ inline void head_bn_rows(float (*d)[128], int nf, const float* g,
                                    const float* b, int tid) {
  if (tid < 32) {
    float m = 0.f;
    for (int j = 0; j < nf; ++j) m += d[tid][j];
    m /= (float)nf;
    float v = 0.f;
    for (int j = 0; j < nf; ++j) {
      float t = d[tid][j] - m;
      v += t * t;
    }
    v /= (float)nf;
    const float rs = rsqrtf(v + 1e-5f) * g[tid];
    const float sh = b[tid];
    for (int j = 0; j < nf; ++j) d[tid][j] = (d[tid][j] - m) * rs + sh;
  }
  __syncthreads();
}

__device__ inline void head_mm(const float (*A)[128], float (*C)[128],
                               const float* W, const float* bias, int NOUT, int K,
                               int tid, bool do_relu) {
  for (int i = tid; i < 32 * NOUT; i += 128) {
    const int r = i / NOUT;
    const int o = i % NOUT;
    float sacc = bias[o];
    for (int k = 0; k < K; ++k) sacc += A[r][k] * W[o * K + k];
    C[r][o] = do_relu ? fmaxf(sacc, 0.f) : sacc;
  }
  __syncthreads();
}

__global__ void __launch_bounds__(128) k_head(
    const float* __restrict__ csum2, const float* __restrict__ bn_g,
    const float* __restrict__ bn_b, const float* w1, const float* b1,
    const float* w2, const float* b2, const float* w22, const float* b22,
    const float* w3, const float* b3, const float* w4, const float* b4,
    const float* w5, const float* b5, float* __restrict__ out) {
  __shared__ float d0[32][128];
  __shared__ float d1[32][128];
  const int tid = threadIdx.x;
  for (int i = tid; i < 32 * 128; i += 128) d0[i >> 7][i & 127] = csum2[i] * 0.25f;
  __syncthreads();
  head_bn_rows(d0, 128, bn_g, bn_b, tid);
  head_mm(d0, d1, w1, b1, 128, 128, tid, true);
  head_mm(d1, d0, w2, b2, 64, 128, tid, true);
  head_bn_rows(d0, 64, bn_g, bn_b, tid);
  head_mm(d0, d1, w22, b22, 64, 64, tid, true);
  head_mm(d1, d0, w3, b3, 32, 64, tid, true);
  head_mm(d0, d1, w4, b4, 32, 32, tid, true);
  head_mm(d1, d0, w5, b5, 16, 32, tid, false);
  if (tid < 32) {
    float sacc = 0.f;
    for (int j = 0; j < 16; ++j) sacc += d0[tid][j];
    out[tid] = fmaxf(sacc, 0.f);
  }
}

// ---------------------------------------------------------------------------
// launcher
// ---------------------------------------------------------------------------
extern "C" void kernel_launch(void* const* d_in, const int* in_sizes, int n_in,
                              void* d_out, int out_size, void* d_ws, size_t ws_size,
                              hipStream_t stream) {
  const int*   atom_ids  = (const int*)d_in[0];
  const int*   batch_idx = (const int*)d_in[3];
  const int*   pos_idx   = (const int*)d_in[4];
  const float* embed_w   = (const float*)d_in[5];
  const float* l1_Wih0 = (const float*)d_in[6];
  const float* l1_Whh0 = (const float*)d_in[7];
  const float* l1_b0   = (const float*)d_in[8];
  const float* l1_Wih  = (const float*)d_in[9];
  const float* l1_Whh  = (const float*)d_in[10];
  const float* l1_b    = (const float*)d_in[11];
  const float* l2_Wih0 = (const float*)d_in[12];
  const float* l2_Whh0 = (const float*)d_in[13];
  const float* l2_b0   = (const float*)d_in[14];
  const float* l2_Wih  = (const float*)d_in[15];
  const float* l2_Whh  = (const float*)d_in[16];
  const float* l2_b    = (const float*)d_in[17];
  const float* bn2_g = (const float*)d_in[18];
  const float* bn2_b = (const float*)d_in[19];
  const float* bn3_g = (const float*)d_in[20];
  const float* bn3_b = (const float*)d_in[21];
  const float* w1 = (const float*)d_in[22];
  const float* b1 = (const float*)d_in[23];
  const float* w2 = (const float*)d_in[24];
  const float* b2 = (const float*)d_in[25];
  const float* w22 = (const float*)d_in[26];
  const float* b22 = (const float*)d_in[27];
  const float* w3 = (const float*)d_in[28];
  const float* b3 = (const float*)d_in[29];
  const float* w4 = (const float*)d_in[30];
  const float* b4 = (const float*)d_in[31];
  const float* w5 = (const float*)d_in[32];
  const float* b5 = (const float*)d_in[33];

  char* ws = (char*)d_ws;
  float*  padded = (float*)(ws);                           // 8 MB
  half_t* x2     = (half_t*)(ws + (size_t)(8)  * 1048576); // 4 MB f16
  half_t* hseqA  = (half_t*)(ws + (size_t)(12) * 1048576); // 8 MB f16
  half_t* hseqB  = (half_t*)(ws + (size_t)(20) * 1048576); // 8 MB f16
  float*  csum2  = (float*)(ws + (size_t)(28) * 1048576);  // 16 KB
  float*  xgT    = (float*)(ws + (size_t)(32) * 1048576);  // 64 MB

  const int N = in_sizes[3];          // 24832 atoms
  const int nblk1 = N / 16;           // 1552

  k_zero<<<2048, 256, 0, stream>>>(padded, (long)32 * 1024 * 64);
  k_zero<<<16, 256, 0, stream>>>(csum2, (long)32 * 128);

  k_lstm1<<<nblk1, 128, 0, stream>>>(atom_ids, embed_w, l1_Wih0, l1_Whh0, l1_b0,
                                     l1_Wih, l1_Whh, l1_b, batch_idx, pos_idx,
                                     padded);

  k_bn2<<<32, 256, 0, stream>>>(padded, bn2_g, bn2_b, x2);

  // LSTM2 layer 0: input = x2 [b][1024][64] (bstr=65536, tstr=64 halfs), K=64
  k_gemm_xg<64><<<2048, 128, 0, stream>>>(x2, 64L, 65536L, l2_Wih0, l2_b0, xgT);
  k_lstm2_rec<<<1, 512, 0, stream>>>(xgT, l2_Whh0, hseqA, csum2, 1);

  // Layers 1..3: input = hseq [t][32][128] (tstr=4096, bstr=128), K=128
  for (int l = 1; l < 4; ++l) {
    const half_t* hin = (l & 1) ? hseqA : hseqB;
    half_t* hout = (l & 1) ? hseqB : hseqA;
    const float* Wih = l2_Wih + (long)(l - 1) * 512 * 128;
    const float* Whh = l2_Whh + (long)(l - 1) * 512 * 128;
    const float* bb  = l2_b   + (long)(l - 1) * 512;
    k_gemm_xg<128><<<2048, 128, 0, stream>>>(hin, 4096L, 128L, Wih, bb, xgT);
    k_lstm2_rec<<<1, 512, 0, stream>>>(xgT, Whh, hout, csum2, (l < 3) ? 1 : 0);
  }

  k_head<<<1, 128, 0, stream>>>(csum2, bn3_g, bn3_b, w1, b1, w2, b2, w22, b22,
                                w3, b3, w4, b4, w5, b5, (float*)d_out);
}
